// TestModel_87359634800816
// MI455X (gfx1250) — matrix operand order repaired and checked by reading, not compiled, not run on hardware
//
#include <hip/hip_runtime.h>

typedef __attribute__((ext_vector_type(16))) _Float16 v16h;
typedef __attribute__((ext_vector_type(2)))  _Float16 h2;
typedef __attribute__((ext_vector_type(8)))  float    v8f;
typedef __attribute__((ext_vector_type(8)))  int      v8i;

#define DIM        32
#define N_LAYERS   82
#define ROW_TILES  4          // 16-row WMMA tiles per wave
#define BATCH      1048576

// lane <-> lane^16 exchange (swap the two 16-lane halves of the wave).
// v_permlanex16_b32 with identity selects; pure VALU, no DS pipe.
__device__ __forceinline__ int permx16(int v) {
#if __has_builtin(__builtin_amdgcn_permlanex16)
    return __builtin_amdgcn_permlanex16(v, v, 0x76543210, 0xFEDCBA98, false, false);
#else
    return __shfl_xor(v, 16, 32);
#endif
}

__device__ __forceinline__ int pack2(float lo, float hi_) {
#if __has_builtin(__builtin_amdgcn_cvt_pkrtz)
    // v_cvt_pk_rtz_f16_f32: lo -> [15:0], hi_ -> [31:16]
    return __builtin_bit_cast(int, __builtin_amdgcn_cvt_pkrtz(lo, hi_));
#else
    h2 p; p.x = (_Float16)lo; p.y = (_Float16)hi_;
    return __builtin_bit_cast(int, p);
#endif
}

// ---------------------------------------------------------------------------
// Prep kernel: pack all 82 layers' weights into a WMMA A-fragment (contiguous k = 16hi+j)
// per-lane layout (f16, 16x32 MxK) and biases into the C/D-fragment layout
// (f32, 8 values per lane).  82 blocks x 64 threads (2 tiles x 32 lanes).
// ---------------------------------------------------------------------------
__global__ void prep_frags(const float* __restrict__ W1, const float* __restrict__ b1,
                           const float* __restrict__ Wm, const float* __restrict__ bm,
                           const float* __restrict__ W6, const float* __restrict__ b6,
                           _Float16* __restrict__ afrag, float* __restrict__ cfrag)
{
    const int l    = blockIdx.x;          // layer 0..81
    const int t    = threadIdx.x >> 5;    // output-feature tile 0..1
    const int lane = threadIdx.x & 31;
    const int Lm   = lane & 15;
    const int hi   = lane >> 4;

    const float* W;
    const float* b;
    if (l == 0)                 { W = W1;                       b = b1; }
    else if (l == N_LAYERS - 1) { W = W6;                       b = b6; }
    else                        { W = Wm + (l - 1) * DIM * DIM; b = bm + (l - 1) * DIM; }

    // hT' = W * hT  ->  A[m,k] = W[16t+m, k]
    const int m = t * 16 + Lm;
    _Float16* ap = afrag + (((size_t)l * 2 + t) * 32 + lane) * 16;
#pragma unroll
    for (int j = 0; j < 16; ++j) {
        int k = hi * 16 + j;
        ap[j] = (_Float16)W[m * DIM + k];
    }

    float* cp = cfrag + (((size_t)l * 2 + t) * 32 + lane) * 8;
#pragma unroll
    for (int r = 0; r < 8; ++r)
        cp[r] = b[t * 16 + (hi ? 8 : 0) + r];
}

// ---------------------------------------------------------------------------
// Main kernel: each wave keeps ROW_TILES B-fragments (activations, packed f16,
// transposed: K=feature, N=batch-row) resident in VGPRs across all 82 layers.
// Per layer: load packed A (weights) + C (bias) fragments, 2*ROW_TILES WMMAs,
// then rebuild next-layer B fragments: cvt_pk to f16 pairs, one permlanex16
// exchange of 4 dwords/tile, cndmask merges.  No LDS, no barriers.
// ---------------------------------------------------------------------------
__global__ __launch_bounds__(256) void mlp_chain82(
    const float*    __restrict__ x,
    float*          __restrict__ out,
    const _Float16* __restrict__ afrag,
    const float*    __restrict__ cfrag)
{
    const int tid  = blockIdx.x * blockDim.x + threadIdx.x;
    const int wave = tid >> 5;
    const int lane = tid & 31;
    const int Lm   = lane & 15;
    const int hi   = lane >> 4;
    const long rowBase = (long)wave * (16 * ROW_TILES);

    const v16h* A = (const v16h*)afrag;   // [layer][tile][lane] -> 16 halves
    const v8f*  C = (const v8f*)cfrag;    // [layer][tile][lane] -> 8 floats

    // ---- load x into B fragments (hT): lane<16: N=Lm, K=0..15; lane>=16: K=16..31
    v8i b[ROW_TILES];
#pragma unroll
    for (int t = 0; t < ROW_TILES; ++t) {
        const float* xp = x + (rowBase + t * 16 + Lm) * DIM + hi * 16;
        v8f p0 = *(const v8f*)(xp);
        v8f p1 = *(const v8f*)(xp + 8);
        v8i bb;
#pragma unroll
        for (int v = 0; v < 4; ++v) {
            bb[v]     = pack2(p0[2 * v], p0[2 * v + 1]);   // k = 16hi + 2v,2v+1
            bb[v + 4] = pack2(p1[2 * v], p1[2 * v + 1]);   // k = 16hi + 8 + 2v,..
        }
        b[t] = bb;
    }

    // ---- layers 0..80: WMMA + rebuild B fragment for next layer
#pragma unroll 1
    for (int l = 0; l < N_LAYERS - 1; ++l) {
        const v16h a0 = A[(size_t)(l * 2 + 0) * 32 + lane];
        const v16h a1 = A[(size_t)(l * 2 + 1) * 32 + lane];
        const v8f  c0 = C[(size_t)(l * 2 + 0) * 32 + lane];
        const v8f  c1 = C[(size_t)(l * 2 + 1) * 32 + lane];
#pragma unroll
        for (int t = 0; t < ROW_TILES; ++t) {
            v8f d0 = __builtin_amdgcn_wmma_f32_16x16x32_f16(
                false, a0, false, __builtin_bit_cast(v16h, b[t]), (short)0, c0, false, false);
            v8f d1 = __builtin_amdgcn_wmma_f32_16x16x32_f16(
                false, a1, false, __builtin_bit_cast(v16h, b[t]), (short)0, c1, false, false);
            // pack f32 D pairs -> f16 dwords (both halves of a B dword come
            // from the same source lane), then exchange only 4 dwords/tile.
            int pk0[4], pk1[4];
#pragma unroll
            for (int v = 0; v < 4; ++v) {
                pk0[v] = pack2(d0[2 * v], d0[2 * v + 1]);  // features  8hi+2v..
                pk1[v] = pack2(d1[2 * v], d1[2 * v + 1]);  // features 16+8hi+2v..
            }
            v8i nb;
#pragma unroll
            for (int v = 0; v < 4; ++v) {
                // send the tile the partner half-wave needs; receive ours
                int z = hi ? pk0[v] : pk1[v];
                int w = permx16(z);
                nb[v]     = hi ? w      : pk0[v];   // k = 16hi + 2v,2v+1
                nb[v + 4] = hi ? pk1[v] : w;        // k = 16hi + 8 + 2v,..
            }
            b[t] = nb;
        }
    }

    // ---- final layer (81): WMMA then store f32 output (un-transposes for free:
    // lane holds 8 consecutive features of one batch row per tile)
    {
        const int l = N_LAYERS - 1;
        const v16h a0 = A[(size_t)(l * 2 + 0) * 32 + lane];
        const v16h a1 = A[(size_t)(l * 2 + 1) * 32 + lane];
        const v8f  c0 = C[(size_t)(l * 2 + 0) * 32 + lane];
        const v8f  c1 = C[(size_t)(l * 2 + 1) * 32 + lane];
#pragma unroll
        for (int t = 0; t < ROW_TILES; ++t) {
            v8f d0 = __builtin_amdgcn_wmma_f32_16x16x32_f16(
                false, a0, false, __builtin_bit_cast(v16h, b[t]), (short)0, c0, false, false);
            v8f d1 = __builtin_amdgcn_wmma_f32_16x16x32_f16(
                false, a1, false, __builtin_bit_cast(v16h, b[t]), (short)0, c1, false, false);
            float* op = out + (rowBase + t * 16 + Lm) * DIM + hi * 8;
            *(v8f*)(op)      = d0;   // features  0..15 block (hi*8 + 0..7)
            *(v8f*)(op + 16) = d1;   // features 16..31 block
        }
    }
}

extern "C" void kernel_launch(void* const* d_in, const int* in_sizes, int n_in,
                              void* d_out, int out_size, void* d_ws, size_t ws_size,
                              hipStream_t stream)
{
    const float* x  = (const float*)d_in[0];
    const float* W1 = (const float*)d_in[1];
    const float* b1 = (const float*)d_in[2];
    const float* Wm = (const float*)d_in[3];
    const float* bm = (const float*)d_in[4];
    const float* W6 = (const float*)d_in[5];
    const float* b6 = (const float*)d_in[6];

    // workspace layout: A fragments (f16) then C fragments (f32), both
    // 82 layers * 2 tiles * 32 lanes; total ~336 KB.
    _Float16* afrag = (_Float16*)d_ws;
    float*    cfrag = (float*)((char*)d_ws +
                               (size_t)N_LAYERS * 2 * 32 * 16 * sizeof(_Float16));

    prep_frags<<<N_LAYERS, 64, 0, stream>>>(W1, b1, Wm, bm, W6, b6, afrag, cfrag);

    const int rows_per_block = 8 * 16 * ROW_TILES;          // 8 waves/block
    const int nblocks = BATCH / rows_per_block;             // 2048
    mlp_chain82<<<nblocks, 256, 0, stream>>>(x, (float*)d_out, afrag, cfrag);
}
